// MultiHeadAttention_26431228739827
// MI455X (gfx1250) — matrix operand order repaired and checked by reading, not compiled, not run on hardware
//
#include <hip/hip_runtime.h>

typedef __bf16 bf16_t;
typedef __bf16 v16bf __attribute__((ext_vector_type(16)));
typedef __bf16 v8bf  __attribute__((ext_vector_type(8)));
typedef float  v8f   __attribute__((ext_vector_type(8)));

#define L_SEQ 2048
#define D_MOD 1024
#define N_TOK 4096
#define N_HEAD 16
#define HD 64

static __device__ __forceinline__ v16bf cat8(v8bf lo, v8bf hi) {
  return __builtin_shufflevector(lo, hi, 0,1,2,3,4,5,6,7,8,9,10,11,12,13,14,15);
}
static __device__ __forceinline__ v8f wmma_bf16(v16bf a, v16bf b, v8f c) {
  return __builtin_amdgcn_wmma_f32_16x16x32_bf16(false, a, false, b, (short)0, c, false, false);
}
static __device__ __forceinline__ v8f v8f_zero() {
  v8f z = {0.f,0.f,0.f,0.f,0.f,0.f,0.f,0.f};
  return z;
}
// CDNA5 async global->LDS copy (ASYNCcnt). LDS address = low 32 bits of the
// flat pointer.
static __device__ __forceinline__ void async_ld_b128(void* lds, const void* g) {
  unsigned int lo = (unsigned int)(size_t)lds;
  asm volatile("global_load_async_to_lds_b128 %0, %1, off"
               :: "v"(lo), "v"(g) : "memory");
}
static __device__ __forceinline__ void wait_async0() {
  asm volatile("s_wait_asynccnt 0x0" ::: "memory");
}

// ---------------------------------------------------------------- convert x (f32 -> bf16)
__global__ void cvt_kernel(const float* __restrict__ s, bf16_t* __restrict__ d, int n) {
  int i = blockIdx.x * blockDim.x + threadIdx.x;
  int stride = gridDim.x * blockDim.x;
  for (; i < n; i += stride) d[i] = (bf16_t)s[i];
}

// ---------------------------------------------------------------- transpose-convert weights
// W[k][n] f32 (1024x1024) -> Wt[n][k] bf16, LDS-tiled 64x64 for coalescing.
__global__ __launch_bounds__(256)
void cvt_tr_kernel(const float* __restrict__ s, bf16_t* __restrict__ d) {
  __shared__ float t[64][65];
  const int k0 = blockIdx.x * 64, n0 = blockIdx.y * 64;
  #pragma unroll
  for (int i = 0; i < 16; ++i) {
    int c = threadIdx.x + i * 256;
    int r = c >> 6, cc = c & 63;
    t[r][cc] = s[(size_t)(k0 + r) * D_MOD + n0 + cc];
  }
  __syncthreads();
  #pragma unroll
  for (int i = 0; i < 16; ++i) {
    int c = threadIdx.x + i * 256;
    int r = c >> 6, cc = c & 63;        // r = local n, cc = local k
    d[(size_t)(n0 + r) * D_MOD + k0 + cc] = (bf16_t)t[cc][r];
  }
}

// ---------------------------------------------------------------- GEMM core
// C[m,n] = sum_k X[m,k] * Wt[n,k]. Block 256 thr (8 waves), tile 128x128, K-step 64,
// double-buffered LDS filled with global_load_async_to_lds_b128.
struct GemmAcc { v8f a[2][4]; };

template <typename EPI>
static __device__ __forceinline__ void gemm_body(const bf16_t* __restrict__ X,
                                                 const bf16_t* __restrict__ Wt,
                                                 EPI epilogue) {
  __shared__ __attribute__((aligned(16))) bf16_t As[2][128][64];
  __shared__ __attribute__((aligned(16))) bf16_t Bs[2][128][64];
  const int tid  = threadIdx.x;
  const int wave = tid >> 5, lane = tid & 31;
  const int lm = lane & 15, lh = lane >> 4;
  const int wm = wave & 3, wn = wave >> 2;       // 4 x 2 wave grid
  const int row0 = blockIdx.x * 128;
  const int col0 = blockIdx.y * 128;

  GemmAcc acc;
  #pragma unroll
  for (int i = 0; i < 2; ++i)
    #pragma unroll
    for (int j = 0; j < 4; ++j) acc.a[i][j] = v8f_zero();

  // issue one K-tile (128x64 of A and of B) as async copies; 4+4 per thread
  auto issue = [&](int k0, int buf) {
    #pragma unroll
    for (int i = 0; i < 4; ++i) {
      int c = tid + i * 256;
      int r = c >> 3, cc = (c & 7) * 8;
      async_ld_b128(&As[buf][r][cc], &X[(size_t)(row0 + r) * D_MOD + k0 + cc]);
    }
    #pragma unroll
    for (int i = 0; i < 4; ++i) {
      int c = tid + i * 256;
      int n = c >> 3, kk = (c & 7) * 8;
      async_ld_b128(&Bs[buf][n][kk], &Wt[(size_t)(col0 + n) * D_MOD + k0 + kk]);
    }
  };

  issue(0, 0);
  wait_async0();
  __syncthreads();

  int cur = 0;
  for (int k0 = 0; k0 < D_MOD; k0 += 64) {
    if (k0 + 64 < D_MOD) issue(k0 + 64, cur ^ 1);
    #pragma unroll
    for (int q = 0; q < 2; ++q) {                 // two K=32 chunks
      v16bf af[2];
      #pragma unroll
      for (int tm = 0; tm < 2; ++tm) {
        const bf16_t* ap = &As[cur][wm * 32 + tm * 16 + lm][q * 32];
        af[tm] = cat8(*(const v8bf*)&ap[8 * lh], *(const v8bf*)&ap[16 + 8 * lh]);
      }
      #pragma unroll
      for (int tn = 0; tn < 4; ++tn) {
        const bf16_t* bp = &Bs[cur][wn * 64 + tn * 16 + lm][q * 32];
        v16bf bfrag = cat8(*(const v8bf*)&bp[8 * lh], *(const v8bf*)&bp[16 + 8 * lh]);
        #pragma unroll
        for (int tm = 0; tm < 2; ++tm)
          acc.a[tm][tn] = wmma_bf16(af[tm], bfrag, acc.a[tm][tn]);
      }
    }
    wait_async0();
    __syncthreads();
    cur ^= 1;
  }
  epilogue(acc, row0, col0, wm, wn, lm, lh);
}

// ---------------------------------------------------------------- QKV projection
// vtrans==0: dst[((b*H+h)*L + l)*HD + hd]   (Q, K layout)
// vtrans==1: dst[((b*H+h)*HD + hd)*L + l]   (V transposed layout)
__global__ __launch_bounds__(256)
void proj_qkv_kernel(const bf16_t* __restrict__ X, const bf16_t* __restrict__ Wt,
                     const float* __restrict__ bias, bf16_t* __restrict__ dst,
                     int vtrans) {
  gemm_body(X, Wt, [&](const GemmAcc& acc, int row0, int col0,
                       int wm, int wn, int lm, int lh) {
    #pragma unroll
    for (int tm = 0; tm < 2; ++tm) {
      #pragma unroll
      for (int tn = 0; tn < 4; ++tn) {
        int n = col0 + wn * 64 + tn * 16 + lm;
        float bv = bias[n];
        int h = n >> 6, hd = n & 63;
        #pragma unroll
        for (int r = 0; r < 8; ++r) {
          int m = row0 + wm * 32 + tm * 16 + r + 8 * lh;
          int b = m >> 11, l = m & 2047;
          size_t idx = vtrans
            ? (((size_t)(b * N_HEAD + h)) * HD + hd) * L_SEQ + l
            : (((size_t)(b * N_HEAD + h)) * L_SEQ + l) * HD + hd;
          dst[idx] = (bf16_t)(acc.a[tm][tn][r] + bv);
        }
      }
    }
  });
}

// ---------------------------------------------------------------- output projection
__global__ __launch_bounds__(256)
void out_proj_kernel(const bf16_t* __restrict__ X, const bf16_t* __restrict__ Wt,
                     const float* __restrict__ bias, float* __restrict__ out) {
  gemm_body(X, Wt, [&](const GemmAcc& acc, int row0, int col0,
                       int wm, int wn, int lm, int lh) {
    #pragma unroll
    for (int tm = 0; tm < 2; ++tm) {
      #pragma unroll
      for (int tn = 0; tn < 4; ++tn) {
        int n = col0 + wn * 64 + tn * 16 + lm;
        float bv = bias[n];
        #pragma unroll
        for (int r = 0; r < 8; ++r) {
          int m = row0 + wm * 32 + tm * 16 + r + 8 * lh;
          out[(size_t)m * D_MOD + n] = acc.a[tm][tn][r] + bv;
        }
      }
    }
  });
}

// ---------------------------------------------------------------- flash attention
// 4 waves/block; each wave independently owns 16 query rows (no barriers).
// Q,K: [B*H, L, HD]; Vt: [B*H, HD, L]; ctx out: [B, L, D], all bf16.
__global__ __launch_bounds__(128)
void attn_kernel(const bf16_t* __restrict__ Q, const bf16_t* __restrict__ K,
                 const bf16_t* __restrict__ Vt, bf16_t* __restrict__ ctx) {
  __shared__ __attribute__((aligned(16))) bf16_t Ps[4][16][32];  // per-wave P tile
  const int tid = threadIdx.x, wave = tid >> 5, lane = tid & 31;
  const int lm = lane & 15, lh = lane >> 4;
  const int bh = blockIdx.y;
  const int b = bh >> 4, head = bh & 15;
  const int qbase = blockIdx.x * 64 + wave * 16;
  const bf16_t* Qp  = Q  + (size_t)bh * L_SEQ * HD;
  const bf16_t* Kp  = K  + (size_t)bh * L_SEQ * HD;
  const bf16_t* Vtp = Vt + (size_t)bh * HD * L_SEQ;

  // Q fragments (A layout), resident for the whole kernel
  v16bf qf[2];
  #pragma unroll
  for (int f = 0; f < 2; ++f) {
    const bf16_t* qp = Qp + (size_t)(qbase + lm) * HD + f * 32;
    qf[f] = cat8(*(const v8bf*)&qp[16 * lh], *(const v8bf*)&qp[16 * lh + 8]);
  }

  v8f acc[4];
  #pragma unroll
  for (int dt = 0; dt < 4; ++dt) acc[dt] = v8f_zero();
  float mi[8], li[8];
  #pragma unroll
  for (int r = 0; r < 8; ++r) { mi[r] = -__builtin_inff(); li[r] = 0.f; }

  const int kend = qbase + 16;                 // per-wave causal bound
  for (int k0 = 0; k0 < kend; k0 += 32) {
    // scores: two 16x16 tiles; K B-frags straight from global (contiguous per lane)
    v8f s[2];
    #pragma unroll
    for (int t = 0; t < 2; ++t) {
      s[t] = v8f_zero();
      int kb = k0 + t * 16;
      #pragma unroll
      for (int f = 0; f < 2; ++f) {
        const bf16_t* kp = Kp + (size_t)(kb + lm) * HD + f * 32 + 16 * lh;
        v16bf kf = cat8(*(const v8bf*)&kp[0], *(const v8bf*)&kp[8]);
        s[t] = wmma_bf16(qf[f], kf, s[t]);
      }
    }
    // online softmax (rows r+8*lh, stats replicated across each 16-lane half)
    #pragma unroll
    for (int r = 0; r < 8; ++r) {
      int qg = qbase + r + 8 * lh;
      float s0 = s[0][r] * 0.125f;
      float s1 = s[1][r] * 0.125f;
      if (k0 + lm > qg)      s0 = -__builtin_inff();
      if (k0 + 16 + lm > qg) s1 = -__builtin_inff();
      float m2 = fmaxf(s0, s1);
      #pragma unroll
      for (int off = 1; off < 16; off <<= 1) m2 = fmaxf(m2, __shfl_xor(m2, off, 32));
      float mnew = fmaxf(mi[r], m2);
      float corr = __expf(mi[r] - mnew);
      float p0 = __expf(s0 - mnew);
      float p1 = __expf(s1 - mnew);
      float rs = p0 + p1;
      #pragma unroll
      for (int off = 1; off < 16; off <<= 1) rs += __shfl_xor(rs, off, 32);
      li[r] = li[r] * corr + rs;
      mi[r] = mnew;
      #pragma unroll
      for (int dt = 0; dt < 4; ++dt) acc[dt][r] *= corr;
      Ps[wave][r + 8 * lh][lm]      = (bf16_t)p0;
      Ps[wave][r + 8 * lh][16 + lm] = (bf16_t)p1;
    }
    // P (A layout) from per-wave LDS; V (B layout) straight from transposed global
    const bf16_t* pp = &Ps[wave][lm][0];
    v16bf pf = cat8(*(const v8bf*)&pp[16 * lh], *(const v8bf*)&pp[16 * lh + 8]);
    #pragma unroll
    for (int dt = 0; dt < 4; ++dt) {
      const bf16_t* vp = Vtp + (size_t)(dt * 16 + lm) * L_SEQ + k0 + 16 * lh;
      v16bf vf = cat8(*(const v8bf*)&vp[0], *(const v8bf*)&vp[8]);
      acc[dt] = wmma_bf16(pf, vf, acc[dt]);
    }
  }

  // epilogue: ctx[b, q, head*64 + d] = acc / li
  #pragma unroll
  for (int dt = 0; dt < 4; ++dt) {
    #pragma unroll
    for (int r = 0; r < 8; ++r) {
      int q = qbase + r + 8 * lh;
      int col = head * HD + dt * 16 + lm;
      float v = acc[dt][r] / li[r];
      ctx[((size_t)(b * L_SEQ + q)) * D_MOD + col] = (bf16_t)v;
    }
  }
}

// ---------------------------------------------------------------- launch
extern "C" void kernel_launch(void* const* d_in, const int* in_sizes, int n_in,
                              void* d_out, int out_size, void* d_ws, size_t ws_size,
                              hipStream_t stream) {
  const float* x  = (const float*)d_in[0];
  const float* Wq = (const float*)d_in[1];
  const float* bq = (const float*)d_in[2];
  const float* Wk = (const float*)d_in[3];
  const float* bk = (const float*)d_in[4];
  const float* Wv = (const float*)d_in[5];
  const float* bv = (const float*)d_in[6];
  const float* Wo = (const float*)d_in[7];
  const float* bo = (const float*)d_in[8];
  float* out = (float*)d_out;

  // workspace layout (bf16 elements)
  bf16_t* ws  = (bf16_t*)d_ws;
  bf16_t* xb  = ws;                  // x bf16           [4096,1024]
  bf16_t* wqb = ws + 4194304;        // Wq^T bf16        [1024,1024]
  bf16_t* wkb = ws + 5242880;        // Wk^T
  bf16_t* wvb = ws + 6291456;        // Wv^T
  bf16_t* wob = ws + 7340032;        // Wo^T
  bf16_t* qb  = ws + 8388608;        // Q  [B,H,L,HD]
  bf16_t* kb  = ws + 12582912;       // K  [B,H,L,HD]
  bf16_t* vb  = ws + 16777216;       // V^T [B,H,HD,L]
  bf16_t* cb  = ws + 20971520;       // ctx [B,L,D]

  cvt_kernel<<<2048, 256, 0, stream>>>(x, xb, N_TOK * D_MOD);
  dim3 gT(16, 16);
  cvt_tr_kernel<<<gT, 256, 0, stream>>>(Wq, wqb);
  cvt_tr_kernel<<<gT, 256, 0, stream>>>(Wk, wkb);
  cvt_tr_kernel<<<gT, 256, 0, stream>>>(Wv, wvb);
  cvt_tr_kernel<<<gT, 256, 0, stream>>>(Wo, wob);

  dim3 gP(N_TOK / 128, D_MOD / 128);
  proj_qkv_kernel<<<gP, 256, 0, stream>>>(xb, wqb, bq, qb, 0);
  proj_qkv_kernel<<<gP, 256, 0, stream>>>(xb, wkb, bk, kb, 0);
  proj_qkv_kernel<<<gP, 256, 0, stream>>>(xb, wvb, bv, vb, 1);

  dim3 gA(L_SEQ / 64, 2 * N_HEAD);
  attn_kernel<<<gA, 128, 0, stream>>>(qb, kb, vb, cb);

  out_proj_kernel<<<gP, 256, 0, stream>>>(cb, wob, bo, out);
}
